// SimpleTransformer_5574867550867
// MI455X (gfx1250) — matrix operand order repaired and checked by reading, not compiled, not run on hardware
//
#include <hip/hip_runtime.h>

// ---------------- WMMA types ----------------
typedef __attribute__((ext_vector_type(16))) __bf16          v16bf;
typedef __attribute__((ext_vector_type(16))) unsigned short  v16us;
typedef __attribute__((ext_vector_type(8)))  float           v8f;

union FragU { v16us u; v16bf b; };

__device__ __forceinline__ unsigned short f2bf(float f) {
    // round-to-nearest-even fp32 -> bf16
    unsigned int u = __float_as_uint(f);
    unsigned int r = u + 0x7FFFu + ((u >> 16) & 1u);
    return (unsigned short)(r >> 16);
}

// LDS byte offset of a generic pointer into __shared__ (aperture keeps it in addr[31:0])
__device__ __forceinline__ unsigned lds_off(const void* p) {
    return (unsigned)(unsigned long long)p;
}

// CDNA5 async global->LDS copy, 16B per lane, tracked by ASYNCcnt (no VGPR data path)
__device__ __forceinline__ void async_copy_b128(unsigned lds, const void* gptr) {
    asm volatile("global_load_async_to_lds_b128 %0, %1, off"
                 :: "v"(lds), "v"((unsigned long long)gptr)
                 : "memory");
}
__device__ __forceinline__ void wait_async0() {
    asm volatile("s_wait_asynccnt 0x0" ::: "memory");
}

// ---------------- Generic bf16 WMMA GEMM ----------------
// C[M,N] = alpha * A[M,K](bf16) @ B + bias (+ReLU).  Per-batch strides via blockIdx.z.
// Block tile 128(M) x 64(N), 4 waves, each wave a 64x32 tile = 8 WMMAs / K-step.
// LDS double-buffered, filled by GLOBAL_LOAD_ASYNC_TO_LDS_B128 (ASYNCcnt pipeline).
// BT:    B given as [N,K] row-major (B^T), else [K,N] row-major.
// BF32:  B source fp32; copied raw to LDS, converted to bf16 at fragment build.
// OBF16: emit bf16 output (chained GEMMs), else fp32.
template<bool BT, bool BF32, bool BIAS, bool RELU, bool OBF16>
__global__ __launch_bounds__(128)
void gemm_wmma(const unsigned short* __restrict__ A, const void* __restrict__ Bv,
               const float* __restrict__ bias, void* __restrict__ Cv,
               int M, int N, int K, float alpha,
               long sAb, long sBb, long sCb)
{
    constexpr int LDA  = 40;   // A row pitch (ushorts): 32 K + pad, 80 B (16B-aligned)
    constexpr int LDBT = 40;   // B^T bf16 [64 n][32 k] row pitch (ushorts)
    constexpr int LDBN = 72;   // B bf16  [32 k][64 n] row pitch (ushorts), 144 B
    constexpr int LDBF = 68;   // B fp32  [32 k][64 n] row pitch (floats), 272 B
    constexpr unsigned SBB = BF32 ? 32u * LDBF * 4u
                                  : (BT ? 64u * LDBT * 2u : 32u * LDBN * 2u);

    __shared__ alignas(16) unsigned short sA[2][128 * LDA];
    __shared__ alignas(16) unsigned char  sBraw[2][SBB];

    const int tid  = threadIdx.x;
    const int lane = tid & 31;
    const int wave = tid >> 5;
    const int wr   = (wave >> 1) * 64;           // wave row offset in 128x64 tile
    const int wc   = (wave & 1)  * 32;           // wave col offset
    const int bm   = blockIdx.y * 128;
    const int bn   = blockIdx.x * 64;
    const long bz  = blockIdx.z;

    const unsigned short* Ab = A + bz * sAb;
    const float*          Bf = (const float*)Bv + (BF32 ? bz * sBb : 0);
    const unsigned short* Bh = (const unsigned short*)Bv + (BF32 ? 0 : bz * sBb);

    const int m   = lane & 15;
    const int khi = lane >> 4;

    v8f acc[4][2] = {};

    // Issue one K-tile's async copies into LDS buffer pb (no VGPR staging).
    auto issue_tiles = [&](int kk, int pb) {
        #pragma unroll
        for (int i = 0; i < 4; ++i) {            // A: 128x32 bf16, 4 x 16B per thread
            int ch = tid + i * 128;
            int r = ch >> 2, c8 = (ch & 3) * 8;
            async_copy_b128(lds_off(&sA[pb][r * LDA + c8]),
                            &Ab[(long)(bm + r) * K + kk + c8]);
        }
        if constexpr (BF32) {
            float* dst = (float*)sBraw[pb];      // [32 k][64 n] fp32
            #pragma unroll
            for (int i = 0; i < 4; ++i) {
                int ch = tid + i * 128;
                int r = ch >> 4, c4 = (ch & 15) * 4;
                async_copy_b128(lds_off(&dst[r * LDBF + c4]),
                                &Bf[(long)(kk + r) * N + bn + c4]);
            }
        } else if constexpr (BT) {
            unsigned short* dst = (unsigned short*)sBraw[pb];   // [64 n][32 k]
            #pragma unroll
            for (int i = 0; i < 2; ++i) {
                int ch = tid + i * 128;
                int r = ch >> 2, c8 = (ch & 3) * 8;
                async_copy_b128(lds_off(&dst[r * LDBT + c8]),
                                &Bh[(long)(bn + r) * K + kk + c8]);
            }
        } else {
            unsigned short* dst = (unsigned short*)sBraw[pb];   // [32 k][64 n]
            #pragma unroll
            for (int i = 0; i < 2; ++i) {
                int ch = tid + i * 128;
                int r = ch >> 3, c8 = (ch & 7) * 8;
                async_copy_b128(lds_off(&dst[r * LDBN + c8]),
                                &Bh[(long)(kk + r) * N + bn + c8]);
            }
        }
    };

    // ---- prologue: stage K-tile 0 ----
    issue_tiles(0, 0);
    wait_async0();
    __syncthreads();
    int p = 0;

    for (int k0 = 0; k0 < K; k0 += 32) {
        const bool nxt = (k0 + 32) < K;
        if (nxt) {
            issue_tiles(k0 + 32, 1 - p);          // async copy overlaps WMMA below
            if (k0 + 64 < K)                      // warm L2 for the tile after that
                __builtin_prefetch(&Ab[(long)(bm + (tid >> 1)) * K + k0 + 64], 0, 1);
        }

        // ---- build fragments (contiguous k = 16*khi + e for A and B) ----
        FragU fa[4], fb[2];
        #pragma unroll
        for (int t = 0; t < 4; ++t)
            #pragma unroll
            for (int e = 0; e < 16; ++e) {
                // A 16x32 bf16: lanes 0-15 = M, lane-half selects K group (k = 16*khi + e)
                int ka = khi * 16 + e;
                fa[t].u[e] = sA[p][(wr + t * 16 + m) * LDA + ka];
            }
        #pragma unroll
        for (int t = 0; t < 2; ++t)
            #pragma unroll
            for (int e = 0; e < 16; ++e) {
                // B 32x16 bf16: lane = N, lane-half selects K group
                int kb = khi * 16 + e;
                int nn = wc + t * 16 + m;
                if constexpr (BF32)
                    fb[t].u[e] = f2bf(((const float*)sBraw[p])[kb * LDBF + nn]);
                else if constexpr (BT)
                    fb[t].u[e] = ((const unsigned short*)sBraw[p])[nn * LDBT + kb];
                else
                    fb[t].u[e] = ((const unsigned short*)sBraw[p])[kb * LDBN + nn];
            }
        #pragma unroll
        for (int i = 0; i < 4; ++i)
            #pragma unroll
            for (int j = 0; j < 2; ++j)
                acc[i][j] = __builtin_amdgcn_wmma_f32_16x16x32_bf16(
                    false, fa[i].b, false, fb[j].b, (short)0, acc[i][j],
                    false, false);

        if (nxt) {
            wait_async0();                        // this wave's copies into 1-p done
            __syncthreads();                      // all waves' copies done / reads done
            p ^= 1;
        }
    }

    // ---- epilogue: C/D layout (VGPR r -> M = r + 8*khi, N = lane&15) ----
    float*          Cf = (float*)Cv + bz * sCb;
    unsigned short* Ch = (unsigned short*)Cv + bz * sCb;
    #pragma unroll
    for (int i = 0; i < 4; ++i) {
        #pragma unroll
        for (int j = 0; j < 2; ++j) {
            #pragma unroll
            for (int r = 0; r < 8; ++r) {
                int row = bm + wr + i * 16 + r + khi * 8;
                int col = bn + wc + j * 16 + (lane & 15);
                float v = acc[i][j][r] * alpha;
                if (BIAS) v += bias[col];
                if (RELU) v = fmaxf(v, 0.0f);
                if (OBF16) Ch[(long)row * N + col] = f2bf(v);
                else       Cf[(long)row * N + col] = v;
            }
        }
    }
}

// ---------------- embedding + positional encoding ----------------
__global__ __launch_bounds__(256)
void embed_k(const int* __restrict__ tokens, const float* __restrict__ emb,
             const float* __restrict__ pe, float* __restrict__ x,
             unsigned short* __restrict__ xb, int S, int D, float scale)
{
    long row = blockIdx.x;               // b*S + s
    int  s   = (int)(row % S);
    int  tok = tokens[row];
    long base = row * (long)D;
    for (int d = threadIdx.x; d < D; d += 256) {
        float v = emb[(long)tok * D + d] * scale + pe[(long)s * D + d];
        x[base + d]  = v;
        xb[base + d] = f2bf(v);
    }
}

// ---------------- causal softmax over S=1024 (in-place fp32 + bf16 copy) ----------------
__global__ __launch_bounds__(256)
void softmax_causal(float* __restrict__ attn, unsigned short* __restrict__ attnb, int S)
{
    __shared__ float sbuf[256];
    const int srow = blockIdx.x;
    const int b    = blockIdx.y;
    const long base = ((long)b * S + srow) * (long)S;
    const int tid  = threadIdx.x;

    float v[4];
    float mx = -1e30f;
    #pragma unroll
    for (int i = 0; i < 4; ++i) {
        int t = tid + i * 256;
        float xv = attn[base + t];
        if (t > srow) xv = -1e9f;        // causal mask
        v[i] = xv;
        mx = fmaxf(mx, xv);
    }
    sbuf[tid] = mx; __syncthreads();
    for (int s = 128; s > 0; s >>= 1) {
        if (tid < s) sbuf[tid] = fmaxf(sbuf[tid], sbuf[tid + s]);
        __syncthreads();
    }
    mx = sbuf[0]; __syncthreads();

    float sum = 0.0f;
    #pragma unroll
    for (int i = 0; i < 4; ++i) { v[i] = __expf(v[i] - mx); sum += v[i]; }
    sbuf[tid] = sum; __syncthreads();
    for (int s = 128; s > 0; s >>= 1) {
        if (tid < s) sbuf[tid] += sbuf[tid + s];
        __syncthreads();
    }
    float inv = 1.0f / sbuf[0]; __syncthreads();

    #pragma unroll
    for (int i = 0; i < 4; ++i) {
        int t = tid + i * 256;
        float o = v[i] * inv;
        attn[base + t]  = o;
        attnb[base + t] = f2bf(o);
    }
}

// ---------------- fused (residual add) + layernorm, emits fp32 + bf16 ----------------
template<bool ADD>
__global__ __launch_bounds__(256)
void add_ln(const float* __restrict__ xin, const float* __restrict__ res,
            const float* __restrict__ g, const float* __restrict__ beta,
            float* __restrict__ xout, unsigned short* __restrict__ xbout, int D)
{
    __shared__ float sbuf[256];
    const long base = (long)blockIdx.x * D;
    const int  tid  = threadIdx.x;

    float v[4];
    float sum = 0.0f;
    #pragma unroll
    for (int i = 0; i < 4; ++i) {
        int d = tid + i * 256;
        float xv = xin[base + d];
        if (ADD) xv += res[base + d];
        v[i] = xv;
        sum += xv;
    }
    sbuf[tid] = sum; __syncthreads();
    for (int s = 128; s > 0; s >>= 1) { if (tid < s) sbuf[tid] += sbuf[tid + s]; __syncthreads(); }
    float mu = sbuf[0] / (float)D; __syncthreads();

    float sq = 0.0f;
    #pragma unroll
    for (int i = 0; i < 4; ++i) { float d0 = v[i] - mu; sq += d0 * d0; }
    sbuf[tid] = sq; __syncthreads();
    for (int s = 128; s > 0; s >>= 1) { if (tid < s) sbuf[tid] += sbuf[tid + s]; __syncthreads(); }
    float rstd = rsqrtf(sbuf[0] / (float)D + 1e-5f); __syncthreads();

    #pragma unroll
    for (int i = 0; i < 4; ++i) {
        int d = tid + i * 256;
        float o = (v[i] - mu) * rstd * g[d] + beta[d];
        xout[base + d]  = o;
        xbout[base + d] = f2bf(o);
    }
}

// ---------------- launch ----------------
extern "C" void kernel_launch(void* const* d_in, const int* in_sizes, int n_in,
                              void* d_out, int out_size, void* d_ws, size_t ws_size,
                              hipStream_t stream)
{
    (void)in_sizes; (void)n_in; (void)out_size; (void)ws_size;
    constexpr int V = 32000, D = 1024, L = 4, S = 1024, Bn = 4, F = 4 * D;

    const int*   tokens = (const int*)d_in[0];
    /* d_in[1] mask: causal by construction, applied analytically */
    const float* emb    = (const float*)d_in[2];
    const float* pe     = (const float*)d_in[3];
    const float* ln1_g  = (const float*)d_in[4];
    const float* ln1_b  = (const float*)d_in[5];
    const float* w1     = (const float*)d_in[6];
    const float* b1     = (const float*)d_in[7];
    const float* w2     = (const float*)d_in[8];
    const float* b2     = (const float*)d_in[9];
    const float* ln2_g  = (const float*)d_in[10];
    const float* ln2_b  = (const float*)d_in[11];
    const float* lnf_g  = (const float*)d_in[12];
    const float* lnf_b  = (const float*)d_in[13];
    const float* headW  = (const float*)d_in[14];
    const float* headb  = (const float*)d_in[15];

    // workspace carve-up (~84 MB)
    char* w = (char*)d_ws;
    float*          x     = (float*)w;          w += (size_t)Bn * S * D * 4;
    float*          tmp   = (float*)w;          w += (size_t)Bn * S * D * 4;
    unsigned short* xb    = (unsigned short*)w; w += (size_t)Bn * S * D * 2;
    unsigned short* attnb = (unsigned short*)w; w += (size_t)Bn * S * S * 2;
    unsigned short* hb    = (unsigned short*)w; w += (size_t)Bn * S * F * 2;

    float* logits    = (float*)d_out;                       // [B,S,V]
    float* attn_base = logits + (size_t)Bn * S * V;         // [L,B,S,S]

    const float scale = 32.0f;                              // sqrt(D)

    embed_k<<<Bn * S, 256, 0, stream>>>(tokens, emb, pe, x, xb, S, D, scale);

    for (int l = 0; l < L; ++l) {
        float* attn_l = attn_base + (size_t)l * Bn * S * S;

        // scores = (x @ x^T) / sqrt(D)   (B^T form: xb is [S,D] row-major)
        gemm_wmma<true, false, false, false, false>
            <<<dim3(S / 64, S / 128, Bn), 128, 0, stream>>>(
                xb, xb, nullptr, attn_l, S, S, D, 1.0f / scale,
                (long)S * D, (long)S * D, (long)S * S);

        softmax_causal<<<dim3(S, Bn), 256, 0, stream>>>(attn_l, attnb, S);

        // attn_out = attn @ x  -> tmp
        gemm_wmma<false, false, false, false, false>
            <<<dim3(D / 64, S / 128, Bn), 128, 0, stream>>>(
                attnb, xb, nullptr, tmp, S, D, S, 1.0f,
                (long)S * S, (long)S * D, (long)S * D);

        add_ln<true><<<Bn * S, 256, 0, stream>>>(x, tmp, ln1_g + (size_t)l * D,
                                                 ln1_b + (size_t)l * D, x, xb, D);

        // h = relu(x @ w1 + b1) -> bf16
        gemm_wmma<false, true, true, true, true>
            <<<dim3(F / 64, (Bn * S) / 128, 1), 128, 0, stream>>>(
                xb, w1 + (size_t)l * D * F, b1 + (size_t)l * F, hb,
                Bn * S, F, D, 1.0f, 0, 0, 0);

        // ffn_out = h @ w2 + b2 -> fp32
        gemm_wmma<false, true, true, false, false>
            <<<dim3(D / 64, (Bn * S) / 128, 1), 128, 0, stream>>>(
                hb, w2 + (size_t)l * F * D, b2 + (size_t)l * D, tmp,
                Bn * S, D, F, 1.0f, 0, 0, 0);

        add_ln<true><<<Bn * S, 256, 0, stream>>>(x, tmp, ln2_g + (size_t)l * D,
                                                 ln2_b + (size_t)l * D, x, xb, D);
    }

    add_ln<false><<<Bn * S, 256, 0, stream>>>(x, nullptr, lnf_g, lnf_b, x, xb, D);

    // logits = x @ headW + headb
    gemm_wmma<false, true, true, false, false>
        <<<dim3(V / 64, (Bn * S) / 128, 1), 128, 0, stream>>>(
            xb, headW, headb, logits, Bn * S, V, D, 1.0f, 0, 0, 0);
}
